// S4Layer_15633680957611
// MI455X (gfx1250) — matrix operand order repaired and checked by reading, not compiled, not run on hardware
//
#include <hip/hip_runtime.h>
#include <hip/hip_bf16.h>

#define PI_F 3.14159265358979323846f

typedef __attribute__((ext_vector_type(16))) _Float16 v16h;
typedef __attribute__((ext_vector_type(8)))  _Float16 v8h;
typedef __attribute__((ext_vector_type(4)))  _Float16 v4h;
typedef __attribute__((ext_vector_type(8)))  float    v8f;

// ---------------- complex helpers ----------------
__device__ __forceinline__ float2 cmul(float2 a, float2 b) {
    return make_float2(a.x * b.x - a.y * b.y, a.x * b.y + a.y * b.x);
}
__device__ __forceinline__ float2 cconj(float2 a) { return make_float2(a.x, -a.y); }
__device__ __forceinline__ float2 cadd(float2 a, float2 b) { return make_float2(a.x + b.x, a.y + b.y); }
__device__ __forceinline__ float2 csub(float2 a, float2 b) { return make_float2(a.x - b.x, a.y - b.y); }
__device__ __forceinline__ float2 cinv(float2 a) {
    float n = a.x * a.x + a.y * a.y;
    return make_float2(a.x / n, -a.y / n);
}

// ---------------- in-LDS radix-2 FFT (256 threads) ----------------
// DIT with explicit bit reversal; inverse = conj twiddles (caller scales by 1/n).
// Twiddle table (if used) is built for base length 4096: tw[j] = exp(-2*pi*i*j/4096).
template <bool USE_TW>
__device__ void fft_lds(float2* __restrict__ s, int n, int logn, bool inverse,
                        const float2* __restrict__ tw) {
    const int tid = threadIdx.x;
    for (int i = tid; i < n; i += 256) {
        unsigned j = __brev((unsigned)i) >> (32 - logn);
        if (j > (unsigned)i) { float2 t = s[i]; s[i] = s[j]; s[j] = t; }
    }
    __syncthreads();
    for (int st = 1; st <= logn; ++st) {
        const int len  = 1 << st;
        const int half = len >> 1;
        const int tstep = 4096 >> st;           // 4096 / len
        const float ang0 = (inverse ? 2.0f : -2.0f) * PI_F / (float)len;
        for (int k = tid; k < (n >> 1); k += 256) {
            const int pos = k & (half - 1);
            const int grp = k >> (st - 1);
            const int i0  = (grp << st) + pos;
            const int i1  = i0 + half;
            float2 w;
            if (USE_TW) {
                w = tw[pos * tstep];
                if (inverse) w.y = -w.y;
            } else {
                float sv, cv;
                __sincosf(ang0 * (float)pos, &sv, &cv);
                w = make_float2(cv, sv);
            }
            float2 a = s[i0];
            float2 b = cmul(s[i1], w);
            s[i0] = cadd(a, b);
            s[i1] = csub(a, b);
        }
        __syncthreads();
    }
}

// ---------------- kernel 0: Wout f32 -> f16 ----------------
__global__ __launch_bounds__(256) void cvt_kernel(const float* __restrict__ w,
                                                  _Float16* __restrict__ wh, int n) {
    int i = blockIdx.x * 256 + threadIdx.x;
    if (i < n) wh[i] = (_Float16)w[i];
}

// ---------------- kernel 1: SSM kernel generation ----------------
// One workgroup per head h. Produces full 4096-bin complex spectrum of the
// bidirectional kernel kk = [k_fwd ; reverse(k_bwd)] into kf_full[h][0..4095].
__global__ __launch_bounds__(256) void kernelgen(
    const float* __restrict__ log_dt, const float* __restrict__ wlr,
    const float* __restrict__ wim,    const float* __restrict__ Pre,
    const float* __restrict__ Pim,    const float* __restrict__ Bre,
    const float* __restrict__ Bim,    const float* __restrict__ Cre,
    const float* __restrict__ Cim,    float2* __restrict__ kf_full) {
    __shared__ float2 S[4096];       // FFT buffer (32 KB)
    __shared__ float  kkb[4096];     // real bidirectional kernel (16 KB)
    __shared__ float2 kf1s[1025];    // saved channel-1 spectrum (8.2 KB)
    __shared__ float2 wds[32];
    __shared__ float2 vs[6][32];
    __shared__ float  dts;

    const int h = blockIdx.x, tid = threadIdx.x;
    if (tid == 0) dts = __expf(log_dt[h]);
    __syncthreads();
    if (tid < 32) {
        const int n = tid, hn = h * 32 + n;
        const float dt = dts;
        float wr = -__expf(wlr[hn]);
        wds[n] = make_float2(wr * dt, wim[hn] * dt);
        float2 Bc  = make_float2(Bre[hn], Bim[hn]);
        float2 Pc  = make_float2(Pre[hn], Pim[hn]);
        float2 C0  = make_float2(Cre[hn], Cim[hn]);
        float2 C1  = make_float2(Cre[8192 + hn], Cim[8192 + hn]);
        float2 Bt_[2] = {Bc, Pc};
        float2 Ct_[3] = {C0, C1, cconj(Pc)};
        #pragma unroll
        for (int a = 0; a < 2; ++a)
            #pragma unroll
            for (int bb = 0; bb < 3; ++bb) vs[a * 3 + bb][n] = cmul(Bt_[a], Ct_[bb]);
    }
    __syncthreads();
    const float dt = dts;

    // Cauchy + Woodbury over Fh = 1025 frequencies.
    for (int f = tid; f <= 1024; f += 256) {
        float sw, cw;
        __sincosf(-2.0f * PI_F * (float)f / 2048.0f, &sw, &cw);
        float2 den = make_float2(1.0f + cw, sw);                   // 1+omega
        float2 num = make_float2(2.0f * (1.0f - cw), -2.0f * sw);  // 2(1-omega)
        float2 zf  = cmul(num, cinv(den));                         // bilinear node z
        float2 r[6];
        #pragma unroll
        for (int c = 0; c < 6; ++c) r[c] = make_float2(0.f, 0.f);
        for (int n = 0; n < 32; ++n) {
            float2 wd  = wds[n];
            float2 d1  = make_float2(zf.x - wd.x, zf.y - wd.y);
            float2 iv1 = cinv(d1);
            float2 d2  = make_float2(zf.x - wd.x, zf.y + wd.y);
            float2 iv2 = cinv(d2);
            #pragma unroll
            for (int c = 0; c < 6; ++c) {
                float2 v = vs[c][n];
                // r += v*iv1 + conj(v)*iv2
                r[c].x += v.x * iv1.x - v.y * iv1.y + v.x * iv2.x + v.y * iv2.y;
                r[c].y += v.x * iv1.y + v.y * iv1.x + v.x * iv2.y - v.y * iv2.x;
            }
        }
        #pragma unroll
        for (int c = 0; c < 6; ++c) { r[c].x *= dt; r[c].y *= dt; }
        // k_f[ch] = r[0,ch] - r[0,2]*r[1,ch]/(1+r[1,2]);  scaled by 2/(1+omega)
        float2 iden = cinv(make_float2(1.0f + r[5].x, r[5].y));
        float2 kf0  = csub(r[0], cmul(r[2], cmul(r[3], iden)));
        float2 kf1  = csub(r[1], cmul(r[2], cmul(r[4], iden)));
        float2 s2   = cinv(den); s2.x *= 2.0f; s2.y *= 2.0f;
        kf0 = cmul(kf0, s2);
        kf1 = cmul(kf1, s2);
        S[f] = kf0;
        if (f > 0 && f < 1024) S[2048 - f] = cconj(kf0);  // Hermitian extension
        kf1s[f] = kf1;
    }
    __syncthreads();

    // irfft channel 0 -> kkb[0..2047]
    fft_lds<false>(S, 2048, 11, true, nullptr);
    for (int t = tid; t < 2048; t += 256) kkb[t] = S[t].x * (1.0f / 2048.0f);
    __syncthreads();

    // irfft channel 1 -> reversed into kkb[2048..4095]
    for (int f = tid; f <= 1024; f += 256) {
        float2 k1 = kf1s[f];
        S[f] = k1;
        if (f > 0 && f < 1024) S[2048 - f] = cconj(k1);
    }
    __syncthreads();
    fft_lds<false>(S, 2048, 11, true, nullptr);
    for (int t = tid; t < 2048; t += 256) kkb[4095 - t] = S[t].x * (1.0f / 2048.0f);
    __syncthreads();

    // forward 4096-pt FFT of kk -> per-head spectrum
    for (int t = tid; t < 4096; t += 256) S[t] = make_float2(kkb[t], 0.0f);
    __syncthreads();
    fft_lds<false>(S, 4096, 12, false, nullptr);
    float2* dst = kf_full + (size_t)h * 4096;
    for (int t = tid; t < 4096; t += 256) dst[t] = S[t];
}

// ---------------- kernel 2: FFT convolution + D-skip + GELU ----------------
// One workgroup per (b,h) row. Output yact f16 in (B,H,L).
__global__ __launch_bounds__(256) void conv_kernel(
    const float* __restrict__ x, const float2* __restrict__ kf_full,
    const float* __restrict__ Dp, _Float16* __restrict__ Yact) {
    __shared__ float2 X[4096];    // 32 KB
    __shared__ float  us[2048];   // 8 KB
    __shared__ float2 tw[2048];   // 16 KB twiddle table (base 4096)

    const int bh = blockIdx.x;
    const int b = bh >> 8, h = bh & 255;
    const int tid = threadIdx.x;

    for (int j = tid; j < 2048; j += 256) {
        float sv, cv;
        __sincosf(-2.0f * PI_F * (float)j / 4096.0f, &sv, &cv);
        tw[j] = make_float2(cv, sv);
    }
    for (int l = tid; l < 2048; l += 256) {
        float v = x[(size_t)l * 4096 + b * 256 + h];   // transpose via L2 (x resident)
        us[l] = v;
        X[l] = make_float2(v, 0.0f);
        X[l + 2048] = make_float2(0.0f, 0.0f);
    }
    __syncthreads();

    fft_lds<true>(X, 4096, 12, false, tw);
    const float2* kf = kf_full + (size_t)h * 4096;
    for (int f = tid; f < 4096; f += 256) X[f] = cmul(X[f], kf[f]);
    __syncthreads();
    fft_lds<true>(X, 4096, 12, true, tw);

    const float Dh = Dp[h];
    _Float16* out = Yact + (size_t)bh * 2048;
    for (int l = tid; l < 2048; l += 256) {
        float y = X[l].x * (1.0f / 4096.0f) + us[l] * Dh;
        float g = 0.5f * y * (1.0f + erff(y * 0.70710678118f));  // exact GELU
        out[l] = (_Float16)g;
    }
}

// ---------------- kernel 3: WMMA GEMM + bias + GLU + residual ----------------
// C(512 x 32cols) = Wout_f16(512x256) * Yact(256 x 32cols of (b,l)).
// 8 waves; wave wv owns M-tiles {wv, wv+8, wv+16, wv+24} for both N-tiles.
// B tiles double-buffered in LDS so staging overlaps the WMMA stream.
__global__ __launch_bounds__(256) void gemm_glu_kernel(
    const _Float16* __restrict__ Wh,   // (512,256) row-major
    const _Float16* __restrict__ Yact, // (B,H,L)
    const float* __restrict__ x,       // (L,B,H) residual source
    const float* __restrict__ bout,    // (512)
    float* __restrict__ z) {           // (L,B,H) pre-LayerNorm output
    __shared__ __align__(16) _Float16 Bt[2][32 * 32];  // [buf][n(32)][k(32)]
    __shared__ float zs[512 * 17];                     // full-M epilogue buffer

    const int l0  = blockIdx.x * 32;
    const int b   = blockIdx.y;
    const int tid = threadIdx.x;
    const int wv  = tid >> 5;          // wave id (wave32)
    const int lane = tid & 31;
    const int Nc  = lane & 15;         // N index within 16-wide tile
    const int hiH = lane >> 4;         // lane-group: K half (B) / M half (C)

    // staging assignment: thread -> (k row, 4 consecutive n cols)
    const int sk = tid >> 3;
    const int sn = (tid & 7) << 2;
    const _Float16* sbase =
        Yact + ((size_t)(b * 256 + sk)) * 2048 + l0 + sn;  // advances by kk0*2048

    v8f acc[2][4];
    #pragma unroll
    for (int t = 0; t < 2; ++t)
        #pragma unroll
        for (int j = 0; j < 4; ++j)
            #pragma unroll
            for (int r = 0; r < 8; ++r) acc[t][j][r] = 0.0f;

    // prologue: stage kk0 = 0 into buffer 0
    {
        v4h d = *(const v4h*)sbase;
        #pragma unroll
        for (int e = 0; e < 4; ++e) Bt[0][(sn + e) * 32 + sk] = d[e];
    }
    __syncthreads();

    for (int s = 0; s < 8; ++s) {
        const int cur = s & 1;
        if (s < 7) {  // stage next K-step into the other buffer (overlaps WMMAs)
            const _Float16* src = sbase + (size_t)(s + 1) * 32 * 2048;
            if (s < 6) __builtin_prefetch(src + (size_t)32 * 2048, 0, 0);
            v4h d = *(const v4h*)src;
            #pragma unroll
            for (int e = 0; e < 4; ++e) Bt[cur ^ 1][(sn + e) * 32 + sk] = d[e];
        }
        // B operands for both N-tiles (contiguous k = 16*hiH + s, same order as A)
        v16h bf0 = *(const v16h*)&Bt[cur][Nc * 32 + hiH * 16];
        v16h bf1 = *(const v16h*)&Bt[cur][(16 + Nc) * 32 + hiH * 16];
        const int kk0 = s * 32;
        #pragma unroll
        for (int j = 0; j < 4; ++j) {
            const int mt = wv + 8 * j;
            const _Float16* arow = Wh + (size_t)(mt * 16 + Nc) * 256 + kk0;
            v8h lo = *(const v8h*)(arow + hiH * 16);
            v8h hi = *(const v8h*)(arow + hiH * 16 + 8);
            v16h af;
            #pragma unroll
            for (int e = 0; e < 8; ++e) { af[e] = lo[e]; af[8 + e] = hi[e]; }
            acc[0][j] = __builtin_amdgcn_wmma_f32_16x16x32_f16(
                false, af, false, bf0, (short)0, acc[0][j], false, false);
            acc[1][j] = __builtin_amdgcn_wmma_f32_16x16x32_f16(
                false, af, false, bf1, (short)0, acc[1][j], false, false);
        }
        __syncthreads();
    }

    // epilogue: two passes (one per N-tile) through the 512x17 LDS buffer
    const int h = tid;
    const float b0 = bout[h], b1 = bout[h + 256];
    #pragma unroll
    for (int t = 0; t < 2; ++t) {
        if (t) __syncthreads();  // pass-0 readers done before overwrite
        #pragma unroll
        for (int j = 0; j < 4; ++j) {
            const int mt = wv + 8 * j;
            #pragma unroll
            for (int r = 0; r < 8; ++r) {
                const int M = mt * 16 + r + hiH * 8;
                zs[M * 17 + Nc] = acc[t][j][r];
            }
        }
        __syncthreads();
        #pragma unroll 4
        for (int c = 0; c < 16; ++c) {
            float a = zs[h * 17 + c] + b0;
            float g = zs[(h + 256) * 17 + c] + b1;
            float glu = a / (1.0f + __expf(-g));
            const size_t idx = (size_t)(l0 + t * 16 + c) * 4096 + b * 256 + h;
            z[idx] = glu + x[idx];
        }
    }
}

// ---------------- kernel 4: LayerNorm over H ----------------
// Row copy uses the CDNA5 async global->LDS path (ASYNCcnt).
__global__ __launch_bounds__(256) void ln_kernel(
    const float* __restrict__ zg, const float* __restrict__ gamma,
    const float* __restrict__ beta, float* __restrict__ out) {
    __shared__ float zrow[256];
    __shared__ float s1[256], s2[256];
    const int row = blockIdx.x, tid = threadIdx.x;

    // async copy: each thread moves its own 4 bytes global -> LDS
    unsigned lds = (unsigned)(uintptr_t)&zrow[tid];
    unsigned long long ga =
        (unsigned long long)(uintptr_t)(zg + (size_t)row * 256 + tid);
    asm volatile("global_load_async_to_lds_b32 %0, %1, off"
                 :: "v"(lds), "v"(ga) : "memory");
    asm volatile("s_wait_asynccnt 0x0" ::: "memory");
    const float v = zrow[tid];

    s1[tid] = v;
    s2[tid] = v * v;
    __syncthreads();
    for (int st = 128; st > 0; st >>= 1) {
        if (tid < st) { s1[tid] += s1[tid + st]; s2[tid] += s2[tid + st]; }
        __syncthreads();
    }
    const float mean = s1[0] * (1.0f / 256.0f);
    const float var  = s2[0] * (1.0f / 256.0f) - mean * mean;
    const float inv  = rsqrtf(var + 1e-5f);
    out[(size_t)row * 256 + tid] = (v - mean) * inv * gamma[tid] + beta[tid];
}

// ---------------- host launcher ----------------
extern "C" void kernel_launch(void* const* d_in, const int* in_sizes, int n_in,
                              void* d_out, int out_size, void* d_ws, size_t ws_size,
                              hipStream_t stream) {
    const float* x       = (const float*)d_in[0];   // (L,B,H)
    const float* log_dt  = (const float*)d_in[1];   // (H,)
    const float* wlr     = (const float*)d_in[2];   // (H,N2)
    const float* wim     = (const float*)d_in[3];   // (H,N2)
    const float* Pre     = (const float*)d_in[4];
    const float* Pim     = (const float*)d_in[5];
    const float* Bre     = (const float*)d_in[6];
    const float* Bim     = (const float*)d_in[7];
    const float* Cre     = (const float*)d_in[8];   // (2,H,N2)
    const float* Cim     = (const float*)d_in[9];
    const float* Dp      = (const float*)d_in[10];  // (1,H)
    const float* Wout    = (const float*)d_in[11];  // (512,256)
    const float* bout    = (const float*)d_in[12];  // (512,)
    const float* gamma   = (const float*)d_in[13];  // (H,)
    const float* beta    = (const float*)d_in[14];  // (H,)
    float* out = (float*)d_out;                     // (L,B,H) f32

    // workspace carve (~56.5 MB)
    char* ws = (char*)d_ws;
    float2*    kf_full = (float2*)ws;                                  //  8 MB
    _Float16*  Yact    = (_Float16*)(ws + (8u << 20));                 // 16 MB
    _Float16*  Wh      = (_Float16*)(ws + (24u << 20));                // 256 KB
    float*     z       = (float*)(ws + (24u << 20) + (1u << 18));      // 32 MB

    cvt_kernel<<<512, 256, 0, stream>>>(Wout, Wh, 512 * 256);
    kernelgen<<<256, 256, 0, stream>>>(log_dt, wlr, wim, Pre, Pim, Bre, Bim,
                                       Cre, Cim, kf_full);
    conv_kernel<<<16 * 256, 256, 0, stream>>>(x, kf_full, Dp, Yact);
    gemm_glu_kernel<<<dim3(2048 / 32, 16), 256, 0, stream>>>(Wh, Yact, x, bout, z);
    ln_kernel<<<2048 * 16, 256, 0, stream>>>(z, gamma, beta, out);
}
